// StickBreakingVAE_48318382080563
// MI455X (gfx1250) — matrix operand order repaired and checked by reading, not compiled, not run on hardware
//
#include <hip/hip_runtime.h>
#include <hip/hip_bf16.h>
#include <cstdint>
#include <cstddef>

// ---------------------------------------------------------------------------
// StickBreakingVAE forward, bf16 WMMA + TDM pipeline for gfx1250 (MI455X)
//   B=32768, D=784 (pad 800), H=500 (pad 512), K=50 (pad 64)
// ---------------------------------------------------------------------------

#define B_SZ 32768
#define D_SZ 784
#define H_SZ 500
#define K_SZ 50
#define DP   800   // D padded to multiple of 32 (K-dim of GEMM1)
#define HP   512   // H padded
#define KP   64    // K padded

typedef __attribute__((ext_vector_type(16))) __bf16          v16bf;
typedef __attribute__((ext_vector_type(8)))  __bf16          v8bf;
typedef __attribute__((ext_vector_type(8)))  unsigned short  v8us;
typedef __attribute__((ext_vector_type(8)))  float           v8f;
typedef __attribute__((ext_vector_type(4)))  unsigned int    u32x4;
typedef __attribute__((ext_vector_type(8)))  int             i32x8;
typedef __attribute__((ext_vector_type(4)))  int             i32x4;

#if defined(__has_builtin)
#if __has_builtin(__builtin_amdgcn_tensor_load_to_lds) && \
    __has_builtin(__builtin_amdgcn_s_wait_tensorcnt)
#define HAVE_TDM 1
#endif
#endif
#ifndef HAVE_TDM
#define HAVE_TDM 0
#endif

__device__ __forceinline__ unsigned short f32_to_bf16_rne(float f) {
  unsigned int u   = __float_as_uint(f);
  unsigned int lsb = (u >> 16) & 1u;
  u += 0x7fffu + lsb;
  return (unsigned short)(u >> 16);
}

enum { ACT_RELU_BF16 = 0, ACT_SOFTPLUS_F32 = 1, ACT_SIGMOID_F32 = 2 };

#if HAVE_TDM
// ---------------------------------------------------------------------------
// TDM: one 2D tile (32 bf16 cols x 128 rows) global -> LDS.
//   - tensor_dim1 = rows_valid: TDM zero-fills LDS for OOB rows (replaces the
//     software Nvalid guard).
//   - LDS padding: pad_interval = 16 DWORDs (one 64B row), pad_amount = 4
//     DWORDs  =>  effective LDS row stride 80B = 40 shorts (bank-conflict pad).
// D# bit layout per CDNA5 ISA 8.3/8.4 (group0 128b, group1 256b, 2D: g2=g3=0).
// This toolchain exposes the 6-arg builtin (extra zero descriptor group).
// ---------------------------------------------------------------------------
__device__ __forceinline__ void tdm_load_tile(const unsigned short* gsrc,
                                              unsigned int lds_byte_off,
                                              unsigned int rows_valid,
                                              unsigned int row_stride_elems) {
  unsigned long long ga = (unsigned long long)(size_t)gsrc;
  u32x4 g0;
  g0[0] = 1u;                                         // count=1 (valid D#)
  g0[1] = lds_byte_off;                               // lds_addr
  g0[2] = (unsigned int)ga;                           // global_addr[31:0]
  g0[3] = (unsigned int)((ga >> 32) & 0x01FFFFFFull)  // global_addr[56:32]
          | 0x80000000u;                              // type=2 ("image")
  i32x8 g1;
  g1[0] = (int)(0x00010000u      // data_size=1 (2 bytes, bf16)
                | (1u << 20)     // pad_enable
                | (3u << 22)     // pad_interval: 16 DWORDs
                | (3u << 25));   // pad_amount:   4 DWORDs
  g1[1] = (int)((32u & 0xFFFFu) << 16);               // tensor_dim0 = 32 (lo16)
  g1[2] = (int)((rows_valid & 0xFFFFu) << 16);        // dim0 hi=0 | tensor_dim1 lo16
  g1[3] = (int)(((rows_valid >> 16) & 0xFFFFu)        // tensor_dim1 hi16
                | (32u << 16));                       // tile_dim0 = 32
  g1[4] = (int)128u;                                  // tile_dim1 = 128, tile_dim2 = 0
  g1[5] = (int)row_stride_elems;                      // tensor_dim0_stride lo32
  g1[6] = 0;                                          // stride hi | dim1_stride lo
  g1[7] = 0;
  i32x4 z4 = {0, 0, 0, 0};
  i32x8 z8 = {0, 0, 0, 0, 0, 0, 0, 0};
  __builtin_amdgcn_tensor_load_to_lds(g0, g1, z4, z4, z8, 0);
}
#endif  // HAVE_TDM

// ---------------------------------------------------------------------------
// x [B,784] f32  ->  x_bf [B,800] bf16 (zero-padded cols)
// ---------------------------------------------------------------------------
__global__ void cvt_x_kernel(const float* __restrict__ x,
                             unsigned short* __restrict__ xb) {
  size_t i = (size_t)blockIdx.x * blockDim.x + threadIdx.x;
  size_t total = (size_t)B_SZ * DP;
  if (i >= total) return;
  int r = (int)(i / DP), c = (int)(i % DP);
  xb[i] = (c < D_SZ) ? f32_to_bf16_rne(x[(size_t)r * D_SZ + c]) : 0;
}

// ---------------------------------------------------------------------------
// W [K,N] f32 -> out [No,Ko] bf16 with out[n][k] = W[k][n]; zero padding
// ---------------------------------------------------------------------------
__global__ void transpose_bf16_kernel(const float* __restrict__ W,
                                      unsigned short* __restrict__ out,
                                      int K, int N, int Ko, int No) {
  int k = blockIdx.x * blockDim.x + threadIdx.x;
  int n = blockIdx.y * blockDim.y + threadIdx.y;
  if (k >= Ko || n >= No) return;
  unsigned short v = 0;
  if (k < K && n < N) v = f32_to_bf16_rne(W[(size_t)k * N + n]);
  out[(size_t)n * Ko + k] = v;
}

// ---------------------------------------------------------------------------
// Tiled bf16 WMMA GEMM:  out = act(A[M x Kd] * Wt^T + bias)
//   A  : bf16 row-major, leading dim ldA, K zero-padded (Kd % 32 == 0)
//   Wt : bf16 W-transposed [Nvalid x ldB] (row n = output column n), K-padded
//   Block: 256 threads = 8 waves; tile 128x128; wave tile 64x32 (4x2 frags)
//   Staging: TDM double-buffered (wave 0 drives the DMA) when available.
// ---------------------------------------------------------------------------
template <int ACT>
__global__ __launch_bounds__(256) void gemm_bf16_wmma_kernel(
    const unsigned short* __restrict__ A,
    const unsigned short* __restrict__ Wt,
    const float* __restrict__ bias,
    void* __restrict__ out,
    int ldA, int Kd, int ldB, int Nvalid, int Nstore, int ldC) {

  const int tid  = threadIdx.x;
  const int wave = tid >> 5;
  const int lane = tid & 31;
  const int lh   = lane >> 4;   // half-wave: 0 or 1
  const int lm   = lane & 15;   // row (A) / col (B,C) within 16

  const int wm = wave & 1;      // 2 waves along M  -> 64 rows each
  const int wn = wave >> 1;     // 4 waves along N  -> 32 cols each

  const int m0 = blockIdx.y * 128;
  const int n0 = blockIdx.x * 128;

  v8f acc[4][2];
#pragma unroll
  for (int mf = 0; mf < 4; ++mf)
#pragma unroll
    for (int nf = 0; nf < 2; ++nf)
#pragma unroll
      for (int r = 0; r < 8; ++r) acc[mf][nf][r] = 0.0f;

#if HAVE_TDM
  // -------- TDM double-buffered pipeline -----------------------------------
  __shared__ unsigned short sA[2][128 * 40];  // row stride 40 shorts (TDM pad)
  __shared__ unsigned short sB[2][128 * 40];

  const unsigned int rowsB =
      (Nvalid > n0) ? (unsigned int)(Nvalid - n0) : 0u;  // OOB rows -> zeros
  const int nsteps = Kd / 32;

  if (wave == 0) {  // prologue: stage step 0 into buffer 0
    tdm_load_tile(A + (size_t)m0 * ldA, (unsigned int)(size_t)&sA[0][0],
                  128u, (unsigned int)ldA);
    tdm_load_tile(Wt + (size_t)n0 * ldB, (unsigned int)(size_t)&sB[0][0],
                  rowsB, (unsigned int)ldB);
    __builtin_amdgcn_s_wait_tensorcnt(0);
  }
  __syncthreads();

  for (int s = 0; s < nsteps; ++s) {
    const int cur = s & 1;
    if (wave == 0 && s + 1 < nsteps) {  // prefetch next tile while computing
      const int k0 = (s + 1) * 32;
      tdm_load_tile(A + (size_t)m0 * ldA + k0,
                    (unsigned int)(size_t)&sA[cur ^ 1][0], 128u,
                    (unsigned int)ldA);
      tdm_load_tile(Wt + (size_t)n0 * ldB + k0,
                    (unsigned int)(size_t)&sB[cur ^ 1][0], rowsB,
                    (unsigned int)ldB);
    }

    // ---- build fragments per ISA VGPR layouts (wave32)
    v16bf a_frag[4];
#pragma unroll
    for (int mf = 0; mf < 4; ++mf) {
      const unsigned short* p =
          &sA[cur][(wm * 64 + mf * 16 + lm) * 40 + 8 * lh];
      v8bf lo = __builtin_bit_cast(v8bf, *(const v8us*)p);        // K = 8*lh + i
      v8bf hi = __builtin_bit_cast(v8bf, *(const v8us*)(p + 16)); // K = 16+8*lh+i
      a_frag[mf] = __builtin_shufflevector(lo, hi, 0, 1, 2, 3, 4, 5, 6, 7,
                                           8, 9, 10, 11, 12, 13, 14, 15);
    }
    v16bf b_frag[2];
#pragma unroll
    for (int nf = 0; nf < 2; ++nf) {
      const unsigned short* p =
          &sB[cur][(wn * 32 + nf * 16 + lm) * 40 + 8 * lh];
      v8bf lo = __builtin_bit_cast(v8bf, *(const v8us*)p);        // K = 8*lh + i
      v8bf hi = __builtin_bit_cast(v8bf, *(const v8us*)(p + 16));
      b_frag[nf] = __builtin_shufflevector(lo, hi, 0, 1, 2, 3, 4, 5, 6, 7,
                                           8, 9, 10, 11, 12, 13, 14, 15);
    }

#pragma unroll
    for (int mf = 0; mf < 4; ++mf)
#pragma unroll
      for (int nf = 0; nf < 2; ++nf)
        acc[mf][nf] = __builtin_amdgcn_wmma_f32_16x16x32_bf16(
            false, a_frag[mf], false, b_frag[nf], (short)0, acc[mf][nf],
            false, false);

    if (wave == 0) __builtin_amdgcn_s_wait_tensorcnt(0);  // next buffer landed
    __syncthreads();  // publish next buffer / retire reads of current
  }
#else
  // -------- fallback: manual staging (single buffer) -----------------------
  __shared__ unsigned short sAm[128][40];
  __shared__ unsigned short sBm[128][40];

  for (int k0 = 0; k0 < Kd; k0 += 32) {
#pragma unroll
    for (int it = 0; it < 2; ++it) {
      int idx = (it * 256 + tid) * 8;  // element index in 128x32 tile
      int r = idx >> 5, c = idx & 31;
      uint4 av = *(const uint4*)(A + (size_t)(m0 + r) * ldA + k0 + c);
      *(uint4*)&sAm[r][c] = av;
      uint4 bv = make_uint4(0u, 0u, 0u, 0u);
      int gn = n0 + r;
      if (gn < Nvalid) bv = *(const uint4*)(Wt + (size_t)gn * ldB + k0 + c);
      *(uint4*)&sBm[r][c] = bv;
    }
    __syncthreads();

    v16bf a_frag[4];
#pragma unroll
    for (int mf = 0; mf < 4; ++mf) {
      const unsigned short* p = &sAm[wm * 64 + mf * 16 + lm][8 * lh];
      v8bf lo = __builtin_bit_cast(v8bf, *(const v8us*)p);
      v8bf hi = __builtin_bit_cast(v8bf, *(const v8us*)(p + 16));
      a_frag[mf] = __builtin_shufflevector(lo, hi, 0, 1, 2, 3, 4, 5, 6, 7,
                                           8, 9, 10, 11, 12, 13, 14, 15);
    }
    v16bf b_frag[2];
#pragma unroll
    for (int nf = 0; nf < 2; ++nf) {
      const unsigned short* p = &sBm[wn * 32 + nf * 16 + lm][8 * lh];
      v8bf lo = __builtin_bit_cast(v8bf, *(const v8us*)p);
      v8bf hi = __builtin_bit_cast(v8bf, *(const v8us*)(p + 16));
      b_frag[nf] = __builtin_shufflevector(lo, hi, 0, 1, 2, 3, 4, 5, 6, 7,
                                           8, 9, 10, 11, 12, 13, 14, 15);
    }

#pragma unroll
    for (int mf = 0; mf < 4; ++mf)
#pragma unroll
      for (int nf = 0; nf < 2; ++nf)
        acc[mf][nf] = __builtin_amdgcn_wmma_f32_16x16x32_bf16(
            false, a_frag[mf], false, b_frag[nf], (short)0, acc[mf][nf],
            false, false);

    __syncthreads();
  }
#endif

  // ---- epilogue: bias + activation, write out
#pragma unroll
  for (int nf = 0; nf < 2; ++nf) {
    int col = n0 + wn * 32 + nf * 16 + lm;
    if (col >= Nstore) continue;
    float bv = (col < Nvalid) ? bias[col] : 0.0f;
#pragma unroll
    for (int mf = 0; mf < 4; ++mf) {
#pragma unroll
      for (int r = 0; r < 8; ++r) {
        int row = m0 + wm * 64 + mf * 16 + r + 8 * lh;  // C/D: M = vgpr + 8*half
        float cv = acc[mf][nf][r] + bv;
        size_t o = (size_t)row * ldC + col;
        if (ACT == ACT_RELU_BF16) {
          ((unsigned short*)out)[o] = f32_to_bf16_rne(fmaxf(cv, 0.0f));
        } else if (ACT == ACT_SOFTPLUS_F32) {
          ((float*)out)[o] = fmaxf(cv, 0.0f) + log1pf(expf(-fabsf(cv)));
        } else {  // ACT_SIGMOID_F32
          ((float*)out)[o] = 1.0f / (1.0f + expf(-cv));
        }
      }
    }
  }
}

// ---------------------------------------------------------------------------
// Kumaraswamy inverse-CDF sample + stick-breaking (serial over K per row)
//   v = (1 - u^(1/beta))^(1/alpha); pi[k] = v[k] * prod_{j<k}(1 - v[j])
// ---------------------------------------------------------------------------
__global__ void stickbreak_kernel(const float* __restrict__ u,
                                  const float* __restrict__ alpha,
                                  const float* __restrict__ beta,
                                  unsigned short* __restrict__ pi_bf) {
  int row = blockIdx.x * blockDim.x + threadIdx.x;
  if (row >= B_SZ) return;
  float stick = 1.0f;
  const size_t base = (size_t)row * K_SZ;
  const size_t obase = (size_t)row * KP;
  for (int k = 0; k < K_SZ; ++k) {
    float a  = alpha[base + k];
    float b  = beta[base + k];
    float uu = u[base + k];
    float t  = fmaxf(1.0f - powf(uu, 1.0f / b), 0.0f);
    float v  = powf(t, 1.0f / a);
    pi_bf[obase + k] = f32_to_bf16_rne(v * stick);
    stick *= (1.0f - v);
  }
  for (int k = K_SZ; k < KP; ++k) pi_bf[obase + k] = 0;
}

// ---------------------------------------------------------------------------
extern "C" void kernel_launch(void* const* d_in, const int* in_sizes, int n_in,
                              void* d_out, int out_size, void* d_ws,
                              size_t ws_size, hipStream_t stream) {
  const float* x       = (const float*)d_in[0];
  const float* u       = (const float*)d_in[1];
  const float* enc_w1  = (const float*)d_in[2];
  const float* enc_b1  = (const float*)d_in[3];
  const float* w_alpha = (const float*)d_in[4];
  const float* b_alpha = (const float*)d_in[5];
  const float* w_beta  = (const float*)d_in[6];
  const float* b_beta  = (const float*)d_in[7];
  const float* dec_w1  = (const float*)d_in[8];
  const float* dec_b1  = (const float*)d_in[9];
  const float* dec_w2  = (const float*)d_in[10];
  const float* dec_b2  = (const float*)d_in[11];

  float* out_recon = (float*)d_out;                                // [B,784]
  float* out_alpha = out_recon + (size_t)B_SZ * D_SZ;              // [B,50]
  float* out_beta  = out_alpha + (size_t)B_SZ * K_SZ;              // [B,50]

  // workspace partition (all bf16 stored as ushort)
  char* w = (char*)d_ws;
  unsigned short* x_bf  = (unsigned short*)w; w += (size_t)B_SZ * DP * 2;
  unsigned short* h_bf  = (unsigned short*)w; w += (size_t)B_SZ * HP * 2;
  unsigned short* hd_bf = (unsigned short*)w; w += (size_t)B_SZ * HP * 2;
  unsigned short* pi_bf = (unsigned short*)w; w += (size_t)B_SZ * KP * 2;
  unsigned short* ew1t  = (unsigned short*)w; w += (size_t)HP * DP * 2;    // [512,800]
  unsigned short* wat   = (unsigned short*)w; w += (size_t)KP * HP * 2;    // [64,512]
  unsigned short* wbt   = (unsigned short*)w; w += (size_t)KP * HP * 2;    // [64,512]
  unsigned short* dw1t  = (unsigned short*)w; w += (size_t)HP * KP * 2;    // [512,64]
  unsigned short* dw2t  = (unsigned short*)w; w += (size_t)D_SZ * HP * 2;  // [784,512]

  // 1) convert/pad x
  {
    size_t total = (size_t)B_SZ * DP;
    cvt_x_kernel<<<(unsigned)((total + 255) / 256), 256, 0, stream>>>(x, x_bf);
  }
  // 2) transpose+convert weights (K,N -> No,Ko)
  {
    dim3 blk(32, 8);
    auto grid = [](int Ko, int No) { return dim3((Ko + 31) / 32, (No + 7) / 8); };
    transpose_bf16_kernel<<<grid(DP, HP), blk, 0, stream>>>(enc_w1, ew1t, D_SZ, H_SZ, DP, HP);
    transpose_bf16_kernel<<<grid(HP, KP), blk, 0, stream>>>(w_alpha, wat, H_SZ, K_SZ, HP, KP);
    transpose_bf16_kernel<<<grid(HP, KP), blk, 0, stream>>>(w_beta,  wbt, H_SZ, K_SZ, HP, KP);
    transpose_bf16_kernel<<<grid(KP, HP), blk, 0, stream>>>(dec_w1, dw1t, K_SZ, H_SZ, KP, HP);
    transpose_bf16_kernel<<<grid(HP, D_SZ), blk, 0, stream>>>(dec_w2, dw2t, H_SZ, D_SZ, HP, D_SZ);
  }

  const dim3 blk(256);
  // 3) h = relu(x @ enc_w1 + b1)        M=B K=800 N=512(valid 500)
  gemm_bf16_wmma_kernel<ACT_RELU_BF16><<<dim3(HP / 128, B_SZ / 128), blk, 0, stream>>>(
      x_bf, ew1t, enc_b1, h_bf, DP, DP, DP, H_SZ, HP, HP);
  // 4) alpha = softplus(h @ w_alpha + b_alpha)   N=50
  gemm_bf16_wmma_kernel<ACT_SOFTPLUS_F32><<<dim3(1, B_SZ / 128), blk, 0, stream>>>(
      h_bf, wat, b_alpha, out_alpha, HP, HP, HP, K_SZ, K_SZ, K_SZ);
  // 5) beta = softplus(h @ w_beta + b_beta)
  gemm_bf16_wmma_kernel<ACT_SOFTPLUS_F32><<<dim3(1, B_SZ / 128), blk, 0, stream>>>(
      h_bf, wbt, b_beta, out_beta, HP, HP, HP, K_SZ, K_SZ, K_SZ);
  // 6) pi via Kumaraswamy sample + stick-breaking
  stickbreak_kernel<<<(B_SZ + 255) / 256, 256, 0, stream>>>(u, out_alpha, out_beta, pi_bf);
  // 7) hd = relu(pi @ dec_w1 + dec_b1)   K=64 N=512(valid 500)
  gemm_bf16_wmma_kernel<ACT_RELU_BF16><<<dim3(HP / 128, B_SZ / 128), blk, 0, stream>>>(
      pi_bf, dw1t, dec_b1, hd_bf, KP, KP, KP, H_SZ, HP, HP);
  // 8) recon = sigmoid(hd @ dec_w2 + dec_b2)  N=784
  gemm_bf16_wmma_kernel<ACT_SIGMOID_F32><<<dim3((D_SZ + 127) / 128, B_SZ / 128), blk, 0, stream>>>(
      hd_bf, dw2t, dec_b2, out_recon, HP, HP, HP, D_SZ, D_SZ, D_SZ);
}
